// Sequence_69630009802910
// MI455X (gfx1250) — matrix operand order repaired and checked by reading, not compiled, not run on hardware
//
#include <hip/hip_runtime.h>
#include <math.h>

typedef __attribute__((ext_vector_type(16))) __bf16    v16bf;
typedef __attribute__((ext_vector_type(8)))  __bf16    v8bf;
typedef __attribute__((ext_vector_type(8)))  float     v8f;
typedef __attribute__((ext_vector_type(8)))  unsigned  v8u;

#define T_STEPS 2048
#define BATCH   256
#define FEAT    64
#define HID     256
#define KDIM    (FEAT + HID)   // 320
#define M_TILE  16             // batch rows per workgroup
#define NTHREADS 256           // 8 waves (wave32), TWO 16-col N-tiles per wave
#define NKT     (KDIM / 32)    // 10 K-tiles of 32

// Opaque packed f32->bf16 convert: result cannot be rematerialized from the
// float sources (pins weight fragments in VGPRs; pins x-conversion placement).
static __device__ __forceinline__ unsigned pk_bf16(float a, float b) {
    unsigned r;
    asm volatile("v_cvt_pk_bf16_f32 %0, %1, %2" : "=v"(r) : "v"(a), "v"(b));
    return r;
}

// Hardware tanh (gfx1250 TRANS op). Trailing v_nop satisfies the TRANS-op
// result-consumption hazard (1 independent op / V_NOP required).
static __device__ __forceinline__ float fast_tanh(float xin) {
    float r;
    asm volatile("v_tanh_f32 %0, %1\n\tv_nop" : "=v"(r) : "v"(xin));
    return r;
}

// Pinned-issue async global load: volatile asm + memory clobber keeps the
// issue point at the top of the step (the WMMA/LDS chain cannot be hoisted
// above it). Result must not be read before wait_all_loads().
static __device__ __forceinline__ float4 load_b128_async(const float* p) {
    float4 r;
    asm volatile("global_load_b128 %0, %1, off" : "=v"(r) : "v"(p) : "memory");
    return r;
}
static __device__ __forceinline__ void wait_all_loads() {
    asm volatile("s_wait_loadcnt 0x0" ::: "memory");
}

union BFrag { v8u u; v16bf b; };
struct U2 { unsigned lo, hi; };

__global__ __launch_bounds__(NTHREADS) __attribute__((amdgpu_waves_per_eu(2)))
void rnn_wmma_kernel(const float* __restrict__ x,
                     const float* __restrict__ W0,
                     const float* __restrict__ b0,
                     const float* __restrict__ Wfc,
                     const float* __restrict__ bfc,
                     float* __restrict__ out)
{
    // double-buffered [x_t | h_{t-1}] tile, bf16, row-major (row = local batch m)
    __shared__ __align__(16) __bf16 comb[2][M_TILE][KDIM];   // 20 KB
    __shared__ float wfc_s[HID];                             // 1 KB

    const int tid = threadIdx.x;
    const int l   = tid & 31;      // lane in wave
    const int w   = tid >> 5;      // wave id 0..7 (owns hidden cols [32w,32w+32))
    const int lh  = l >> 4;        // half-wave 0/1
    const int ll  = l & 15;
    const int b_base = blockIdx.x * M_TILE;

    // ---------------- init ----------------
    wfc_s[tid] = Wfc[tid];

    // h_0 = 0 in buffer 0
    for (int i = tid; i < M_TILE * HID; i += NTHREADS)
        comb[0][i >> 8][FEAT + (i & 255)] = (__bf16)0.0f;

    // x_0 into buffer 0 (256 threads x float4)
    const int xm = (tid * 4) >> 6;       // local batch row this thread loads
    const int xf = (tid * 4) & 63;       // feature offset
    {
        const float4 xv = *(const float4*)(x + ((size_t)(b_base + xm)) * FEAT + xf);
        U2 p; p.lo = pk_bf16(xv.x, xv.y); p.hi = pk_bf16(xv.z, xv.w);
        *(U2*)&comb[0][xm][xf] = p;
    }

    // ---- preload this wave's W0^T B-fragments into registers (packed bf16) ----
    // B is 32x16 (KxN): tile0 col n0 = 32w + ll, tile1 col n1 = n0 + 16,
    // K = kt*32 + lh*16 + j
    const int ncol0 = w * 32 + ll;
    const int ncol1 = ncol0 + 16;
    v8u bfrag0[NKT], bfrag1[NKT];
    #pragma unroll
    for (int kt = 0; kt < NKT; ++kt) {
        const float* wr0 = W0 + (size_t)ncol0 * KDIM + kt * 32 + lh * 16;
        const float* wr1 = W0 + (size_t)ncol1 * KDIM + kt * 32 + lh * 16;
        v8u b0f, b1f;
        #pragma unroll
        for (int j = 0; j < 8; ++j) {
            b0f[j] = pk_bf16(wr0[2 * j], wr0[2 * j + 1]);
            b1f[j] = pk_bf16(wr1[2 * j], wr1[2 * j + 1]);
        }
        bfrag0[kt] = b0f;
        bfrag1[kt] = b1f;
    }
    const float b0v0 = b0[ncol0];
    const float b0v1 = b0[ncol1];
    const float bfcv = bfc[0];

    __syncthreads();

    // out-reduction mapping: row rm, 16-wide hidden chunk rj
    const int rm = tid >> 4;
    const int rj = tid & 15;

    // ---------------- sequential steps ----------------
    for (int t = 0; t < T_STEPS; ++t) {
        const int cur = t & 1, nxt = cur ^ 1;

        // (1) issue x_{t+1} global load NOW (pinned by volatile asm); it has
        //     the whole WMMA chain + fc1 reduction to cover its latency.
        //     Also warm L2 for x_{t+2} (global_prefetch: no counters, no regs),
        //     so next step's load is a cache hit well inside the wait window.
        const bool do_x = (t + 1 < T_STEPS);
        float4 xv = {0.0f, 0.0f, 0.0f, 0.0f};
        if (do_x)
            xv = load_b128_async(x + ((size_t)(t + 1) * BATCH + b_base + xm) * FEAT + xf);
        if (t + 2 < T_STEPS)
            __builtin_prefetch(x + ((size_t)(t + 2) * BATCH + b_base + xm) * FEAT + xf, 0, 3);

        // (2) GEMM chain: 20 x v_wmma_f32_16x16x32_bf16 (2 N-tiles).
        //     One A-fragment pair feeds both tiles; A loads double-buffered.
        v8f c0, c1;
        #pragma unroll
        for (int r = 0; r < 8; ++r) { c0[r] = b0v0; c1[r] = b0v1; }

        v8bf alo[2], ahi[2];
        alo[0] = *(const v8bf*)&comb[cur][ll][lh * 16];
        ahi[0] = *(const v8bf*)&comb[cur][ll][lh * 16 + 8];
        #pragma unroll
        for (int kt = 0; kt < NKT; ++kt) {
            const int cb = kt & 1, nb = cb ^ 1;
            if (kt + 1 < NKT) {
                const int kb = (kt + 1) * 32 + lh * 16;
                alo[nb] = *(const v8bf*)&comb[cur][ll][kb];
                ahi[nb] = *(const v8bf*)&comb[cur][ll][kb + 8];
            }
            v16bf a;
            #pragma unroll
            for (int j = 0; j < 8; ++j) { a[j] = alo[cb][j]; a[j + 8] = ahi[cb][j]; }
            BFrag f0; f0.u = bfrag0[kt];
            BFrag f1; f1.u = bfrag1[kt];
            c0 = __builtin_amdgcn_wmma_f32_16x16x32_bf16(
                     false, a, false, f0.b, (short)0, c0, false, false);
            c1 = __builtin_amdgcn_wmma_f32_16x16x32_bf16(
                     false, a, false, f1.b, (short)0, c1, false, false);
        }

        // (3) fc1 output for t-1 (reads h_{t-1} in comb[cur]); overlaps both
        //     the WMMA drain and the in-flight x load.
        if (t > 0) {
            float acc = 0.0f;
            #pragma unroll
            for (int n = 0; n < 16; ++n)
                acc += (float)comb[cur][rm][FEAT + rj * 16 + n] * wfc_s[rj * 16 + n];
            acc += __shfl_xor(acc, 1);
            acc += __shfl_xor(acc, 2);
            acc += __shfl_xor(acc, 4);
            acc += __shfl_xor(acc, 8);
            if (rj == 0) out[(size_t)(b_base + rm) * T_STEPS + (t - 1)] = acc + bfcv;
        }

        // (4) land x_{t+1} into comb[nxt]: explicit wait (only loadcnt event
        //     in the loop), then packed cvt + 8-byte LDS store.
        if (do_x) {
            wait_all_loads();
            U2 p; p.lo = pk_bf16(xv.x, xv.y); p.hi = pk_bf16(xv.z, xv.w);
            *(U2*)&comb[nxt][xm][xf] = p;
        }

        // (5) tanh + store h_t (bf16); C/D layout: elem r -> M = lh*8+r
        #pragma unroll
        for (int r = 0; r < 8; ++r) {
            const float h0 = fast_tanh(c0[r]);
            comb[nxt][lh * 8 + r][FEAT + ncol0] = (__bf16)h0;
            const float h1 = fast_tanh(c1[r]);
            comb[nxt][lh * 8 + r][FEAT + ncol1] = (__bf16)h1;
        }

        __syncthreads();
    }

    // final output column: h_{T-1} sits in comb[T_STEPS & 1]
    {
        const int cur = T_STEPS & 1;
        float acc = 0.0f;
        #pragma unroll
        for (int n = 0; n < 16; ++n)
            acc += (float)comb[cur][rm][FEAT + rj * 16 + n] * wfc_s[rj * 16 + n];
        acc += __shfl_xor(acc, 1);
        acc += __shfl_xor(acc, 2);
        acc += __shfl_xor(acc, 4);
        acc += __shfl_xor(acc, 8);
        if (rj == 0) out[(size_t)(b_base + rm) * T_STEPS + (T_STEPS - 1)] = acc + bfcv;
    }
}

extern "C" void kernel_launch(void* const* d_in, const int* in_sizes, int n_in,
                              void* d_out, int out_size, void* d_ws, size_t ws_size,
                              hipStream_t stream) {
    const float* x   = (const float*)d_in[0];
    const float* W0  = (const float*)d_in[1];
    const float* b0  = (const float*)d_in[2];
    const float* Wfc = (const float*)d_in[3];
    const float* bfc = (const float*)d_in[4];
    float* out = (float*)d_out;
    // 16 workgroups (one per 16-row batch slice), 256 threads = 8 waves each
    hipLaunchKernelGGL(rnn_wmma_kernel,
                       dim3(BATCH / M_TILE), dim3(NTHREADS), 0, stream,
                       x, W0, b0, Wfc, bfc, out);
}
